// NoisyTopkRouter_20426864459935
// MI455X (gfx1250) — compile-verified, matrix operand order checked by reading, not run on hardware
//
#include <hip/hip_runtime.h>

typedef __attribute__((ext_vector_type(16))) __bf16 v16bf;
typedef __attribute__((ext_vector_type(8)))  float  v8f;
typedef __attribute__((ext_vector_type(4)))  float  v4f;

#define NEG_INF (-1e30f)

static constexpr int  Bsz    = 8;
static constexpr int  S      = 4096;
static constexpr int  D      = 1024;
static constexpr int  E      = 64;
static constexpr long ROWS   = (long)Bsz * S;          // 32768
static constexpr int  KSTEPS = D / 32;                 // 32
static constexpr long ROUTE_OFF = 0;
static constexpr long IX_OFF    = ROWS * E;            // 2097152
static constexpr long FULL_OFF  = IX_OFF + ROWS * 2;   // 2162688

// ---------------------------------------------------------------------------
// Pre-kernel: convert Wl/Wn (f32, [E=64, D=1024], row-major) into bf16 WMMA
// B-fragments in workspace.
// Layout: wf[((m*4 + c)*32 + t)*512 + lane*16 + i]   (bf16 elements)
//   m    : 0=Wl, 1=Wn
//   c    : expert col-tile (16 experts each)
//   t    : K step (32 K-values each)
//   lane : wave lane (B 32x16 bf16 layout: lane n<16 holds col n, K 0..7 &
//          16..23 of the step; lane n+16 holds col n, K 8..15 & 24..31)
//   i    : element within the lane's v16bf fragment
// ---------------------------------------------------------------------------
__global__ void cvt_weights_kernel(const float* __restrict__ Wl,
                                   const float* __restrict__ Wn,
                                   __bf16* __restrict__ wf) {
  int tid = blockIdx.x * blockDim.x + threadIdx.x;   // 131072 threads total
  int m    = tid >> 16;
  int c    = (tid >> 14) & 3;
  int t    = (tid >> 9)  & 31;
  int lane = (tid >> 4)  & 31;
  int i    = tid & 15;
  int expert = c * 16 + (lane & 15);
  int k = t * 32 + ((lane >> 4) ? 8 : 0) + (i < 8 ? i : i + 8);
  const float* W = (m == 0) ? Wl : Wn;
  wf[tid] = (__bf16)W[expert * D + k];
}

__device__ __forceinline__ float softplus_f(float x) {
  // stable softplus: max(x,0) + log1p(exp(-|x|))
  return fmaxf(x, 0.0f) + log1pf(__expf(-fabsf(x)));
}

// ---------------------------------------------------------------------------
// Main fused kernel: dual GEMM (logits + noise-std linear) via bf16 WMMA,
// then softplus/noise combine, full softmax, top-2, sparse softmax.
// One wave = one 16-row tile x all 64 experts x both matrices.
// ---------------------------------------------------------------------------
__global__ __launch_bounds__(256)
void router_main_kernel(const float*  __restrict__ h,
                        const float*  __restrict__ bl,
                        const float*  __restrict__ bn,
                        const float*  __restrict__ noise,
                        const __bf16* __restrict__ wf,
                        float*        __restrict__ out) {
  __shared__ float lds[8][16][65];   // 65: pad to avoid bank conflicts

  const int lane = threadIdx.x & 31;
  const int wave = threadIdx.x >> 5;
  const int half = lane >> 4;        // 0 or 1
  const int mr   = lane & 15;        // A-matrix row within tile
  const long tile = (long)blockIdx.x * 8 + wave;
  const long row0 = tile * 16;

  v8f accL[4], accN[4];
#pragma unroll
  for (int c = 0; c < 4; ++c) {
#pragma unroll
    for (int r = 0; r < 8; ++r) { accL[c][r] = 0.0f; accN[c][r] = 0.0f; }
  }

  // A-fragment base: lane n<16 reads row (row0+n) bytes [0..31] & [64..95] of
  // each 128B K-step segment; lane n+16 reads [32..63] & [96..127].
  const float* arow = h + (row0 + mr) * D + half * 8;

  for (int t = 0; t < KSTEPS; ++t) {
    const float* p = arow + t * 32;
    v4f a0 = *(const v4f*)(p +  0);
    v4f a1 = *(const v4f*)(p +  4);
    v4f a2 = *(const v4f*)(p + 16);
    v4f a3 = *(const v4f*)(p + 20);
    v16bf afrag;
#pragma unroll
    for (int j = 0; j < 4; ++j) {
      afrag[j     ] = (__bf16)a0[j];
      afrag[j +  4] = (__bf16)a1[j];
      afrag[j +  8] = (__bf16)a2[j];
      afrag[j + 12] = (__bf16)a3[j];
    }

#pragma unroll
    for (int c = 0; c < 4; ++c) {
      v16bf bL = *(const v16bf*)(wf + ((long)( c     * 32 + t) * 512) + lane * 16);
      v16bf bN = *(const v16bf*)(wf + ((long)((4 + c) * 32 + t) * 512) + lane * 16);
      accL[c] = __builtin_amdgcn_wmma_f32_16x16x32_bf16(
          false, afrag, false, bL, (short)0, accL[c], false, false);
      accN[c] = __builtin_amdgcn_wmma_f32_16x16x32_bf16(
          false, afrag, false, bN, (short)0, accN[c], false, false);
    }
  }

  // Epilogue part 1: bias, softplus, noise combine -> noisy logits into LDS.
  // C/D layout: VGPR r, lanes 0-15: (M=r,   N=lane), lanes 16-31: (M=r+8, N=lane-16)
  const int eIdx = lane & 15;
#pragma unroll
  for (int c = 0; c < 4; ++c) {
    const int   e   = c * 16 + eIdx;
    const float blv = bl[e];
    const float bnv = bn[e];
#pragma unroll
    for (int r = 0; r < 8; ++r) {
      const int  rm  = half * 8 + r;
      const long row = row0 + rm;
      float lg = accL[c][r] + blv;
      float sp = softplus_f(accN[c][r] + bnv);
      float nz = noise[row * E + e];
      lds[wave][rm][e] = lg + nz * sp;
    }
  }
  __syncthreads();

  // Epilogue part 2: per row (all 32 lanes cooperate; 2 experts per lane).
  for (int r = 0; r < 16; ++r) {
    const long row = row0 + r;
    float v0 = lds[wave][r][lane];
    float v1 = lds[wave][r][lane + 32];

    // full softmax
    float mx = fmaxf(v0, v1);
#pragma unroll
    for (int off = 16; off > 0; off >>= 1) mx = fmaxf(mx, __shfl_xor(mx, off, 32));
    float e0 = __expf(v0 - mx);
    float e1 = __expf(v1 - mx);
    float s  = e0 + e1;
#pragma unroll
    for (int off = 16; off > 0; off >>= 1) s += __shfl_xor(s, off, 32);
    float inv = 1.0f / s;
    out[FULL_OFF + row * E + lane]      = e0 * inv;
    out[FULL_OFF + row * E + lane + 32] = e1 * inv;

    // argmax (ties -> lowest index, matching lax.top_k)
    float bv; int bi;
    if (v0 >= v1) { bv = v0; bi = lane; } else { bv = v1; bi = lane + 32; }
#pragma unroll
    for (int off = 16; off > 0; off >>= 1) {
      float ov = __shfl_xor(bv, off, 32);
      int   oi = __shfl_xor(bi, off, 32);
      if (ov > bv || (ov == bv && oi < bi)) { bv = ov; bi = oi; }
    }
    // second max with winner masked out
    float v0m = (bi == lane     ) ? NEG_INF : v0;
    float v1m = (bi == lane + 32) ? NEG_INF : v1;
    float bv2; int bi2;
    if (v0m >= v1m) { bv2 = v0m; bi2 = lane; } else { bv2 = v1m; bi2 = lane + 32; }
#pragma unroll
    for (int off = 16; off > 0; off >>= 1) {
      float ov = __shfl_xor(bv2, off, 32);
      int   oi = __shfl_xor(bi2, off, 32);
      if (ov > bv2 || (ov == bv2 && oi < bi2)) { bv2 = ov; bi2 = oi; }
    }

    // sparse (top-2) softmax; non-top-k entries are exactly 0 (exp(-1e30-m)->0)
    float rinv = 1.0f / (1.0f + __expf(bv2 - bv));
    float r0 = (lane      == bi || lane      == bi2) ? __expf(v0 - bv) * rinv : 0.0f;
    float r1 = (lane + 32 == bi || lane + 32 == bi2) ? __expf(v1 - bv) * rinv : 0.0f;
    out[ROUTE_OFF + row * E + lane]      = r0;
    out[ROUTE_OFF + row * E + lane + 32] = r1;

    if (lane == 0) {
      out[IX_OFF + row * 2]     = (float)bi;
      out[IX_OFF + row * 2 + 1] = (float)bi2;
    }
  }
}

extern "C" void kernel_launch(void* const* d_in, const int* in_sizes, int n_in,
                              void* d_out, int out_size, void* d_ws, size_t ws_size,
                              hipStream_t stream) {
  const float* h     = (const float*)d_in[0];
  const float* Wl    = (const float*)d_in[1];
  const float* bl    = (const float*)d_in[2];
  const float* Wn    = (const float*)d_in[3];
  const float* bn    = (const float*)d_in[4];
  const float* noise = (const float*)d_in[5];
  // d_in[6] = top_k (fixed at 2 in this kernel)
  (void)in_sizes; (void)n_in; (void)out_size; (void)ws_size;

  __bf16* wf = (__bf16*)d_ws;  // 2*4*32*32*16 bf16 = 256 KB

  // 131072 weight fragment elements
  cvt_weights_kernel<<<131072 / 256, 256, 0, stream>>>(Wl, Wn, wf);

  // 32768 rows / (16 rows/wave * 8 waves/block) = 256 blocks
  router_main_kernel<<<(int)(ROWS / (16 * 8)), 256, 0, stream>>>(
      h, bl, bn, noise, wf, (float*)d_out);
}
